// AttnBlock_54752243089536
// MI455X (gfx1250) — matrix operand order repaired and checked by reading, not compiled, not run on hardware
//
#include <hip/hip_runtime.h>

// ---------------------------------------------------------------------------
// Types for CDNA5 WMMA (wave32)
// ---------------------------------------------------------------------------
typedef __attribute__((ext_vector_type(16))) __bf16    v16bf;
typedef __attribute__((ext_vector_type(8)))  float     v8f;
typedef __attribute__((ext_vector_type(4)))  unsigned  v4u;

#define WMMA_BF16(A, Bf, Cacc) \
    __builtin_amdgcn_wmma_f32_16x16x32_bf16(false, (A), false, (Bf), (short)0, (Cacc), false, false)

// Problem constants
#define BATCH 32
#define NTOK  1024      // H*W
#define CCH   512
#define NGRP  32
#define CPG   16        // C / G

__device__ __forceinline__ unsigned short f32_to_bf16(float f) {
    union { float f; unsigned u; } a; a.f = f;
    unsigned u = a.u;
    u += 0x7FFFu + ((u >> 16) & 1u);        // round-to-nearest-even
    return (unsigned short)(u >> 16);
}

// Load a v16bf fragment from two 16-byte-aligned 16B chunks.
__device__ __forceinline__ v16bf load_frag(const unsigned short* p0,
                                           const unsigned short* p1) {
    union { v4u u[2]; v16bf v; } x;
    x.u[0] = *(const v4u*)p0;
    x.u[1] = *(const v4u*)p1;
    return x.v;
}

// Convert 16 fp32 (two v8f) into a bf16 A/B fragment.
__device__ __forceinline__ v16bf cvt_frag(v8f lo, v8f hi) {
    union { unsigned short u[16]; v16bf v; } x;
#pragma unroll
    for (int i = 0; i < 8; ++i) {
        x.u[i]     = f32_to_bf16(lo[i]);
        x.u[8 + i] = f32_to_bf16(hi[i]);
    }
    return x.v;
}

// ---------------------------------------------------------------------------
// Kernel 0: pack a 512x512 row-major fp32 weight (K x N) into WMMA-B bf16
// layout: [kt(16)][nt(32)][lane(32)][e(16)] -> one contiguous 32B load/lane.
// grid=(512, 4 matrices), block=32 (one wave).
// ---------------------------------------------------------------------------
__global__ __launch_bounds__(32) void pack_weights(
    const float* __restrict__ w0, const float* __restrict__ w1,
    const float* __restrict__ w2, const float* __restrict__ w3,
    unsigned short* __restrict__ dst) {
    const float* W = (blockIdx.y == 0) ? w0 : (blockIdx.y == 1) ? w1
                   : (blockIdx.y == 2) ? w2 : w3;
    int tile = blockIdx.x;
    int kt = tile >> 5, nt = tile & 31;
    int lane = threadIdx.x;
    int n = lane & 15, h = lane >> 4;
    unsigned short* d = dst + (size_t)blockIdx.y * (CCH * CCH)
                      + ((size_t)(kt * 32 + nt) * 32 + lane) * 16;
#pragma unroll
    for (int e = 0; e < 16; ++e) {
        int K = kt * 32 + 16 * h + e;                 // B-layout: K = 16h + e
        d[e] = f32_to_bf16(W[K * CCH + nt * 16 + n]);
    }
}

// ---------------------------------------------------------------------------
// Kernel 1: GroupNorm over (H,W,C/G) per (batch, group) -> bf16 hn (B,N,C).
// x values cached in registers (64/thread) so HBM is read once.
// grid = B*G = 1024, block = 256.
// ---------------------------------------------------------------------------
__global__ __launch_bounds__(256) void gn_kernel(
    const float* __restrict__ x, const float* __restrict__ scale,
    const float* __restrict__ bias, unsigned short* __restrict__ hn) {
    __shared__ float ssum[256], ssq[256];
    int b = blockIdx.x >> 5, g = blockIdx.x & 31;
    float xv[64];
    float s = 0.f, sq = 0.f;
#pragma unroll
    for (int j = 0; j < 64; ++j) {
        int i = threadIdx.x + j * 256;
        int t = i >> 4;
        int c = g * CPG + (i & 15);
        float v = x[((size_t)b * NTOK + t) * CCH + c];
        xv[j] = v;
        s += v; sq += v * v;
    }
    ssum[threadIdx.x] = s; ssq[threadIdx.x] = sq;
    __syncthreads();
    for (int off = 128; off > 0; off >>= 1) {
        if ((int)threadIdx.x < off) {
            ssum[threadIdx.x] += ssum[threadIdx.x + off];
            ssq[threadIdx.x]  += ssq[threadIdx.x + off];
        }
        __syncthreads();
    }
    float mean = ssum[0] * (1.0f / (NTOK * CPG));
    float var  = ssq[0]  * (1.0f / (NTOK * CPG)) - mean * mean;
    float rinv = rsqrtf(var + 1e-6f);
#pragma unroll
    for (int j = 0; j < 64; ++j) {
        int i = threadIdx.x + j * 256;
        int t = i >> 4;
        int c = g * CPG + (i & 15);
        float r = (xv[j] - mean) * rinv * scale[c] + bias[c];
        hn[((size_t)b * NTOK + t) * CCH + c] = f32_to_bf16(r);
    }
}

// ---------------------------------------------------------------------------
// Kernel 2: GEMM  out(R,512) = A(R,512) @ Wpk + bias   (bf16 in, bf16 out)
// Wave = 16-row strip x full 512 cols. A strip cached in 128 VGPRs.
// Two N-tiles per iteration (shared A) + software-pipelined B fragments so
// WMMAs overlap the next fragment's loads (partial s_wait_loadcnt).
// transposeOut=1 stores per-batch (C,N) transposed (for V).
// grid = (B*N/16)/4 = 512, block = 128 (4 waves).
// ---------------------------------------------------------------------------
__global__ __launch_bounds__(128) void gemm_bf16(
    const unsigned short* __restrict__ A,
    const unsigned short* __restrict__ Wpk,
    const float* __restrict__ bias,
    unsigned short* __restrict__ out, int transposeOut) {
    int lane = threadIdx.x & 31;
    int wv   = threadIdx.x >> 5;
    int rowTile = blockIdx.x * 4 + wv;
    int m = lane & 15, h = lane >> 4, n = m;

    v16bf a[16];
    {
        const unsigned short* base = A + (size_t)(rowTile * 16 + m) * CCH;
#pragma unroll
        for (int ks = 0; ks < 16; ++ks) {
            const unsigned short* p = base + ks * 32 + 16 * h;
            a[ks] = load_frag(p, p + 8);
        }
    }
    for (int ntp = 0; ntp < 16; ++ntp) {
        int nt0 = ntp * 2, nt1 = nt0 + 1;
        // K-step stride in packed layout = 32*32*16 = 16384 elements
        const unsigned short* B0 = Wpk + ((size_t)nt0 * 32 + lane) * 16;
        const unsigned short* B1 = Wpk + ((size_t)nt1 * 32 + lane) * 16;
        v16bf bc0 = load_frag(B0, B0 + 8);
        v16bf bc1 = load_frag(B1, B1 + 8);
        v8f acc0 = {}, acc1 = {};
#pragma unroll
        for (int ks = 0; ks < 16; ++ks) {
            v16bf bn0 = bc0, bn1 = bc1;
            if (ks < 15) {                       // issue next frags before WMMA
                const unsigned short* p0 = B0 + (size_t)(ks + 1) * 16384;
                const unsigned short* p1 = B1 + (size_t)(ks + 1) * 16384;
                bn0 = load_frag(p0, p0 + 8);
                bn1 = load_frag(p1, p1 + 8);
            }
            acc0 = WMMA_BF16(a[ks], bc0, acc0);
            acc1 = WMMA_BF16(a[ks], bc1, acc1);
            bc0 = bn0; bc1 = bn1;
        }
        float bb0 = bias[nt0 * 16 + n];
        float bb1 = bias[nt1 * 16 + n];
        if (!transposeOut) {
#pragma unroll
            for (int r = 0; r < 8; ++r) {
                int row = rowTile * 16 + r + 8 * h;
                out[(size_t)row * CCH + nt0 * 16 + n] = f32_to_bf16(acc0[r] + bb0);
                out[(size_t)row * CCH + nt1 * 16 + n] = f32_to_bf16(acc1[r] + bb1);
            }
        } else {
            int batch = (rowTile * 16) >> 10;    // 16-row tile never crosses a batch
            size_t ob = (size_t)batch * CCH * NTOK;
#pragma unroll
            for (int r = 0; r < 8; ++r) {
                int row = rowTile * 16 + r + 8 * h;
                int tok = row & (NTOK - 1);
                out[ob + (size_t)(nt0 * 16 + n) * NTOK + tok] = f32_to_bf16(acc0[r] + bb0);
                out[ob + (size_t)(nt1 * 16 + n) * NTOK + tok] = f32_to_bf16(acc1[r] + bb1);
            }
        }
    }
}

// ---------------------------------------------------------------------------
// Kernel 3: attention for a 64-row q tile of one batch.
// Full 64x1024 fp32 logits in 256KB LDS (CDNA5: 320KB/WGP).
// grid = B * (N/64) = 512, block = 128 (4 waves, 16 q-rows each).
// ---------------------------------------------------------------------------
__global__ __launch_bounds__(128) void attn_kernel(
    const unsigned short* __restrict__ q, const unsigned short* __restrict__ k,
    const unsigned short* __restrict__ vT, unsigned short* __restrict__ o) {
    extern __shared__ float smem[];   // 64 * 1024 floats = 256 KB
    __shared__ float redbuf[128];
    int lane = threadIdx.x & 31;
    int wv   = threadIdx.x >> 5;
    int b    = blockIdx.x >> 4;
    int qBase = (blockIdx.x & 15) * 64;
    int m = lane & 15, h = lane >> 4, n = m;
    const float sc = 0.04419417382415922f;   // 1/sqrt(512)

    // Cache this wave's 16 q-rows as A fragments (128 VGPRs).
    v16bf a[16];
    {
        const unsigned short* base = q + (size_t)(b * NTOK + qBase + wv * 16 + m) * CCH;
#pragma unroll
        for (int ks = 0; ks < 16; ++ks) {
            const unsigned short* p = base + ks * 32 + 16 * h;
            a[ks] = load_frag(p, p + 8);
        }
    }
    // Phase 1: logits = q @ k^T * sc -> LDS (fp32). Two KV-tiles/iter, shared A,
    // pipelined B fragments.
    for (int ktp = 0; ktp < 32; ++ktp) {
        int kt0 = ktp * 2, kt1 = kt0 + 1;
        const unsigned short* K0 =
            k + (size_t)(b * NTOK + kt0 * 16 + n) * CCH + 16 * h;
        const unsigned short* K1 = K0 + (size_t)16 * CCH;
        v16bf bc0 = load_frag(K0, K0 + 8);
        v16bf bc1 = load_frag(K1, K1 + 8);
        v8f acc0 = {}, acc1 = {};
#pragma unroll
        for (int ks = 0; ks < 16; ++ks) {
            v16bf bn0 = bc0, bn1 = bc1;
            if (ks < 15) {
                const unsigned short* p0 = K0 + (ks + 1) * 32;
                const unsigned short* p1 = K1 + (ks + 1) * 32;
                bn0 = load_frag(p0, p0 + 8);
                bn1 = load_frag(p1, p1 + 8);
            }
            acc0 = WMMA_BF16(a[ks], bc0, acc0);
            acc1 = WMMA_BF16(a[ks], bc1, acc1);
            bc0 = bn0; bc1 = bn1;
        }
#pragma unroll
        for (int r = 0; r < 8; ++r) {
            smem[(wv * 16 + r + 8 * h) * NTOK + kt0 * 16 + n] = acc0[r] * sc;
            smem[(wv * 16 + r + 8 * h) * NTOK + kt1 * 16 + n] = acc1[r] * sc;
        }
    }
    __syncthreads();
    // Phase 2: softmax. Two threads per row (all 128 threads active).
    {
        int row  = threadIdx.x >> 1;
        int half = threadIdx.x & 1;
        float* rowp = smem + (size_t)row * NTOK + half * 512;
        float mx = -3.4e38f;
        for (int i = 0; i < 512; ++i) mx = fmaxf(mx, rowp[i]);
        redbuf[threadIdx.x] = mx;
        __syncthreads();
        mx = fmaxf(redbuf[row * 2], redbuf[row * 2 + 1]);
        float sum = 0.f;
        for (int i = 0; i < 512; ++i) {
            float e = __expf(rowp[i] - mx);
            rowp[i] = e; sum += e;
        }
        __syncthreads();
        redbuf[threadIdx.x] = sum;
        __syncthreads();
        float inv = 1.0f / (redbuf[row * 2] + redbuf[row * 2 + 1]);
        for (int i = 0; i < 512; ++i) rowp[i] *= inv;
    }
    __syncthreads();
    // Phase 3: o = P @ v. Two N-tiles/iter share the converted A fragment
    // (halves LDS reads + cvt work); pipelined B fragments from transposed v.
    for (int ntp = 0; ntp < 16; ++ntp) {
        int nt0 = ntp * 2, nt1 = nt0 + 1;
        const unsigned short* V0 = vT + (size_t)b * CCH * NTOK
                                 + (size_t)(nt0 * 16 + n) * NTOK + 16 * h;
        const unsigned short* V1 = V0 + (size_t)16 * NTOK;
        v16bf bc0 = load_frag(V0, V0 + 8);
        v16bf bc1 = load_frag(V1, V1 + 8);
        v8f acc0 = {}, acc1 = {};
#pragma unroll 4
        for (int kt2 = 0; kt2 < 32; ++kt2) {
            v16bf bn0 = bc0, bn1 = bc1;
            if (kt2 < 31) {
                const unsigned short* p0 = V0 + (kt2 + 1) * 32;
                const unsigned short* p1 = V1 + (kt2 + 1) * 32;
                bn0 = load_frag(p0, p0 + 8);
                bn1 = load_frag(p1, p1 + 8);
            }
            const float* ap = smem + (size_t)(wv * 16 + m) * NTOK + kt2 * 32 + 16 * h;
            v16bf af = cvt_frag(*(const v8f*)ap, *(const v8f*)(ap + 8));
            acc0 = WMMA_BF16(af, bc0, acc0);
            acc1 = WMMA_BF16(af, bc1, acc1);
            bc0 = bn0; bc1 = bn1;
        }
#pragma unroll
        for (int r = 0; r < 8; ++r) {
            int row = b * NTOK + qBase + wv * 16 + r + 8 * h;
            o[(size_t)row * CCH + nt0 * 16 + n] = f32_to_bf16(acc0[r]);
            o[(size_t)row * CCH + nt1 * 16 + n] = f32_to_bf16(acc1[r]);
        }
    }
}

// ---------------------------------------------------------------------------
// Kernel 4: out = x + o @ Wp + bp   (fp32 output, residual)
// Same pairing/pipelining as kernel 2. grid = 512, block = 128.
// ---------------------------------------------------------------------------
__global__ __launch_bounds__(128) void proj_res_kernel(
    const unsigned short* __restrict__ o, const unsigned short* __restrict__ Wpk,
    const float* __restrict__ bias, const float* __restrict__ x,
    float* __restrict__ out) {
    int lane = threadIdx.x & 31;
    int wv   = threadIdx.x >> 5;
    int rowTile = blockIdx.x * 4 + wv;
    int m = lane & 15, h = lane >> 4, n = m;

    v16bf a[16];
    {
        const unsigned short* base = o + (size_t)(rowTile * 16 + m) * CCH;
#pragma unroll
        for (int ks = 0; ks < 16; ++ks) {
            const unsigned short* p = base + ks * 32 + 16 * h;
            a[ks] = load_frag(p, p + 8);
        }
    }
    for (int ntp = 0; ntp < 16; ++ntp) {
        int nt0 = ntp * 2, nt1 = nt0 + 1;
        const unsigned short* B0 = Wpk + ((size_t)nt0 * 32 + lane) * 16;
        const unsigned short* B1 = Wpk + ((size_t)nt1 * 32 + lane) * 16;
        v16bf bc0 = load_frag(B0, B0 + 8);
        v16bf bc1 = load_frag(B1, B1 + 8);
        v8f acc0 = {}, acc1 = {};
#pragma unroll
        for (int ks = 0; ks < 16; ++ks) {
            v16bf bn0 = bc0, bn1 = bc1;
            if (ks < 15) {
                const unsigned short* p0 = B0 + (size_t)(ks + 1) * 16384;
                const unsigned short* p1 = B1 + (size_t)(ks + 1) * 16384;
                bn0 = load_frag(p0, p0 + 8);
                bn1 = load_frag(p1, p1 + 8);
            }
            acc0 = WMMA_BF16(a[ks], bc0, acc0);
            acc1 = WMMA_BF16(a[ks], bc1, acc1);
            bc0 = bn0; bc1 = bn1;
        }
        float bb0 = bias[nt0 * 16 + n];
        float bb1 = bias[nt1 * 16 + n];
#pragma unroll
        for (int r = 0; r < 8; ++r) {
            size_t i0 = (size_t)(rowTile * 16 + r + 8 * h) * CCH + nt0 * 16 + n;
            size_t i1 = (size_t)(rowTile * 16 + r + 8 * h) * CCH + nt1 * 16 + n;
            out[i0] = x[i0] + acc0[r] + bb0;
            out[i1] = x[i1] + acc1[r] + bb1;
        }
    }
}

// ---------------------------------------------------------------------------
// Launch
// ---------------------------------------------------------------------------
extern "C" void kernel_launch(void* const* d_in, const int* in_sizes, int n_in,
                              void* d_out, int out_size, void* d_ws, size_t ws_size,
                              hipStream_t stream) {
    (void)in_sizes; (void)n_in; (void)out_size; (void)ws_size;
    const float* x  = (const float*)d_in[0];
    const float* gs = (const float*)d_in[1];
    const float* gb = (const float*)d_in[2];
    const float* Wq = (const float*)d_in[3];
    const float* bq = (const float*)d_in[4];
    const float* Wk = (const float*)d_in[5];
    const float* bk = (const float*)d_in[6];
    const float* Wv = (const float*)d_in[7];
    const float* bv = (const float*)d_in[8];
    const float* Wp = (const float*)d_in[9];
    const float* bp = (const float*)d_in[10];

    char* ws = (char*)d_ws;
    const size_t TOK_BYTES = (size_t)BATCH * NTOK * CCH * 2;   // 32 MB each
    unsigned short* hn  = (unsigned short*)(ws);
    unsigned short* q   = (unsigned short*)(ws + 1 * TOK_BYTES);
    unsigned short* k   = (unsigned short*)(ws + 2 * TOK_BYTES);
    unsigned short* vT  = (unsigned short*)(ws + 3 * TOK_BYTES);
    unsigned short* o   = (unsigned short*)(ws + 4 * TOK_BYTES);
    unsigned short* wpk = (unsigned short*)(ws + 5 * TOK_BYTES); // 4 x 512KB
    unsigned short* wq_pk = wpk + 0 * (CCH * CCH);
    unsigned short* wk_pk = wpk + 1 * (CCH * CCH);
    unsigned short* wv_pk = wpk + 2 * (CCH * CCH);
    unsigned short* wp_pk = wpk + 3 * (CCH * CCH);

    pack_weights<<<dim3(16 * 32, 4), 32, 0, stream>>>(Wq, Wk, Wv, Wp, wpk);
    gn_kernel<<<BATCH * NGRP, 256, 0, stream>>>(x, gs, gb, hn);

    const int gemmBlocks = (BATCH * NTOK / 16) / 4;   // 512
    gemm_bf16<<<gemmBlocks, 128, 0, stream>>>(hn, wq_pk, bq, q, 0);
    gemm_bf16<<<gemmBlocks, 128, 0, stream>>>(hn, wk_pk, bk, k, 0);
    gemm_bf16<<<gemmBlocks, 128, 0, stream>>>(hn, wv_pk, bv, vT, 1);

    attn_kernel<<<BATCH * (NTOK / 64), 128, 64 * NTOK * sizeof(float), stream>>>(q, k, vT, o);

    proj_res_kernel<<<gemmBlocks, 128, 0, stream>>>(o, wp_pk, bp, x, (float*)d_out);
}
